// GCN_68710886801557
// MI455X (gfx1250) — matrix operand order repaired and checked by reading, not compiled, not run on hardware
//
#include <hip/hip_runtime.h>

// ---------------------------------------------------------------------------
// GCN forward for MI455X (gfx1250, wave32).
// Dominant cost: edge scatter (~3.5 GB gather/scatter across 4 layers) ->
// bandwidth bound at 23.3 TB/s; global_atomic_add_f32, one wave per edge.
// Dense GEMMs (~10 GFLOP) run on v_wmma_f32_16x16x32_bf16 (bf16 in / f32
// accumulate) with branch-free vectorized fragment loads.
// ---------------------------------------------------------------------------

#define HIDC 128
#define NGRAPH 64

typedef __attribute__((ext_vector_type(16))) __bf16 v16bf;
typedef __attribute__((ext_vector_type(8)))  float  v8f;

// ------------------------------- utility -----------------------------------

__global__ __launch_bounds__(256) void zero_f32(float* p, long long n) {
  long long i = (long long)blockIdx.x * blockDim.x + threadIdx.x;
  long long stride = (long long)gridDim.x * blockDim.x;
  for (; i < n; i += stride) p[i] = 0.0f;
}

__global__ __launch_bounds__(256) void deg_init(float* deg, int n) {
  int i = blockIdx.x * 256 + threadIdx.x;
  if (i < n) deg[i] = 1.0f;                      // self-loop contributes 1
}

__global__ __launch_bounds__(256) void deg_accum(const long long* ei, float* deg, int E) {
  int e = blockIdx.x * 256 + threadIdx.x;
  if (e < E) atomicAdd(&deg[(int)ei[(long long)E + e]], 1.0f);  // row 1 = dst
}

__global__ __launch_bounds__(256) void deg_inv(const float* deg, float* dinv, int n) {
  int i = blockIdx.x * 256 + threadIdx.x;
  if (i < n) dinv[i] = rsqrtf(deg[i]);
}

// --------------------- layer-0 GEMM (K=4, bandwidth bound) -----------------
// out[n,128] = x[n,4] @ W0[4,128]; one thread per 4 output channels.
__global__ __launch_bounds__(256) void gemm_k4(
    const float* __restrict__ x, const float* __restrict__ W0,
    float* __restrict__ out, int n) {
  long long t = (long long)blockIdx.x * 256 + threadIdx.x;   // n*32 threads
  int node = (int)(t >> 5);
  if (node >= n) return;
  int q = ((int)t & 31) * 4;
  float4 xv = *(const float4*)(x + (size_t)node * 4);
  float4 w0 = *(const float4*)(W0 + 0 * HIDC + q);
  float4 w1 = *(const float4*)(W0 + 1 * HIDC + q);
  float4 w2 = *(const float4*)(W0 + 2 * HIDC + q);
  float4 w3 = *(const float4*)(W0 + 3 * HIDC + q);
  float4 o;
  o.x = xv.x * w0.x + xv.y * w1.x + xv.z * w2.x + xv.w * w3.x;
  o.y = xv.x * w0.y + xv.y * w1.y + xv.z * w2.y + xv.w * w3.y;
  o.z = xv.x * w0.z + xv.y * w1.z + xv.z * w2.z + xv.w * w3.z;
  o.w = xv.x * w0.w + xv.y * w1.w + xv.z * w2.w + xv.w * w3.w;
  *(float4*)(out + (size_t)node * HIDC + q) = o;
}

// ------------------------------ WMMA GEMM ----------------------------------
// C[M,N] = A[M,K] @ B[K,N] (+ bias[N]).  fp32 memory, bf16 WMMA, f32 acc.
// REQUIRES: M % 16 == 0, N % 16 == 0, K % 32 == 0 (all call sites satisfy).
// One wave per 16x16 tile; branch-free loop body: 4x b128 A loads,
// 16x b32 B loads, pack to bf16, one v_wmma per chunk.  Tile guard is
// wave-uniform -> EXEC all-1s at every WMMA (ISA 7.12 requirement).
__global__ __launch_bounds__(256) void wmma_gemm_bf16(
    const float* __restrict__ A, const float* __restrict__ B,
    const float* __restrict__ bias, float* __restrict__ C,
    int M, int N, int K) {
  int lane = threadIdx.x & 31;
  int wave = threadIdx.x >> 5;
  int ntiles = N >> 4;
  int totalTiles = (M >> 4) * ntiles;
  int tile = blockIdx.x * (blockDim.x >> 5) + wave;
  if (tile >= totalTiles) return;                // uniform across the wave
  int mt = tile / ntiles;
  int nt = tile - mt * ntiles;
  int m0 = mt << 4, n0 = nt << 4;
  int col = lane & 15;                            // A row / B,C column index
  int hi  = lane >> 4;                            // lane half select

  // A 16x32 fragment per chunk: lane half 0 holds K = 0..15,
  // half 1 holds K = 16..31  -> one contiguous 16-float segment.
  const float* Arow = A + (size_t)(m0 + col) * K + hi * 16;
  // B 32x16 fragment: half 0 holds K = 0..15, half 1 K = 16..31 (col n0+col)
  const float* Bcol = B + (size_t)(hi * 16) * N + n0 + col;

  v8f acc = {};
  for (int kc = 0; kc < K; kc += 32) {
    float4 a0 = *(const float4*)(Arow + kc);
    float4 a1 = *(const float4*)(Arow + kc + 4);
    float4 a2 = *(const float4*)(Arow + kc + 8);
    float4 a3 = *(const float4*)(Arow + kc + 12);
    v16bf af, bf;
    af[0]  = (__bf16)a0.x; af[1]  = (__bf16)a0.y;
    af[2]  = (__bf16)a0.z; af[3]  = (__bf16)a0.w;
    af[4]  = (__bf16)a1.x; af[5]  = (__bf16)a1.y;
    af[6]  = (__bf16)a1.z; af[7]  = (__bf16)a1.w;
    af[8]  = (__bf16)a2.x; af[9]  = (__bf16)a2.y;
    af[10] = (__bf16)a2.z; af[11] = (__bf16)a2.w;
    af[12] = (__bf16)a3.x; af[13] = (__bf16)a3.y;
    af[14] = (__bf16)a3.z; af[15] = (__bf16)a3.w;
    const float* bp = Bcol + (size_t)kc * N;
    #pragma unroll
    for (int j = 0; j < 16; ++j)
      bf[j] = (__bf16)bp[(size_t)j * N];
    acc = __builtin_amdgcn_wmma_f32_16x16x32_bf16(
        false, af, false, bf, (short)0, acc, false, false);
  }

  float bv = bias ? bias[n0 + col] : 0.0f;
  #pragma unroll
  for (int r = 0; r < 8; ++r) {
    int rr = r + hi * 8;                          // C VGPR r: M = r + 8*hi
    C[(size_t)(m0 + rr) * N + n0 + col] = acc[r] + bv;
  }
}

// --------------------------- edge scatter ----------------------------------
// One wave per edge; lane q handles channels 4q..4q+3 (float4 gather,
// 4x global_atomic_add_f32 scatter).  Self-loops are folded into gcn_post.
__global__ __launch_bounds__(256) void scatter_edges(
    const long long* __restrict__ ei, const float* __restrict__ hW,
    const float* __restrict__ dinv, float* __restrict__ agg, int E) {
  long long tid = (long long)blockIdx.x * blockDim.x + threadIdx.x;
  long long e = tid >> 5;
  if (e >= E) return;
  int q = (int)(tid & 31);
  int s = (int)ei[e];                             // row 0 = src
  int d = (int)ei[(long long)E + e];              // row 1 = dst
  float nrm = dinv[s] * dinv[d];
  const float4 hv = *(const float4*)(hW + (size_t)s * HIDC + q * 4);
  float* ap = agg + (size_t)d * HIDC + q * 4;
  atomicAdd(ap + 0, hv.x * nrm);
  atomicAdd(ap + 1, hv.y * nrm);
  atomicAdd(ap + 2, hv.z * nrm);
  atomicAdd(ap + 3, hv.w * nrm);
}

// ------------------- self-loop + bias + ReLU + LayerNorm -------------------
// One wave per node (4 channels/lane), wave32 shfl_xor reductions.
__global__ __launch_bounds__(256) void gcn_post(
    const float* __restrict__ agg, const float* __restrict__ hW,
    const float* __restrict__ dinv, const float* __restrict__ bias,
    const float* __restrict__ g, const float* __restrict__ be,
    float* __restrict__ out, int n) {
  int node = blockIdx.x * (blockDim.x >> 5) + (threadIdx.x >> 5);
  if (node >= n) return;
  int lane = threadIdx.x & 31;
  float di = dinv[node];
  float self = di * di;                           // self-loop norm
  float v[4];
  float s = 0.0f;
  #pragma unroll
  for (int i = 0; i < 4; ++i) {
    int c = lane * 4 + i;
    float x = agg[(size_t)node * HIDC + c] +
              hW[(size_t)node * HIDC + c] * self + bias[c];
    x = fmaxf(x, 0.0f);                           // ReLU before LN
    v[i] = x;
    s += x;
  }
  #pragma unroll
  for (int o = 16; o >= 1; o >>= 1) s += __shfl_xor(s, o, 32);
  float mu = s * (1.0f / HIDC);
  float vs = 0.0f;
  #pragma unroll
  for (int i = 0; i < 4; ++i) { float d2 = v[i] - mu; vs += d2 * d2; }
  #pragma unroll
  for (int o = 16; o >= 1; o >>= 1) vs += __shfl_xor(vs, o, 32);
  float rs = rsqrtf(vs * (1.0f / HIDC) + 1e-5f);
  #pragma unroll
  for (int i = 0; i < 4; ++i) {
    int c = lane * 4 + i;
    out[(size_t)node * HIDC + c] = (v[i] - mu) * rs * g[c] + be[c];
  }
}

// ----------------------------- mean pool -----------------------------------

__global__ __launch_bounds__(256) void pool_accum(
    const float* __restrict__ h, const long long* __restrict__ batch,
    float* __restrict__ featsum, float* __restrict__ cnt, int n) {
  int node = blockIdx.x * (blockDim.x >> 5) + (threadIdx.x >> 5);
  if (node >= n) return;
  int lane = threadIdx.x & 31;
  int gidx = (int)batch[node];
  #pragma unroll
  for (int i = 0; i < 4; ++i) {
    int c = lane * 4 + i;
    atomicAdd(&featsum[(size_t)gidx * HIDC + c], h[(size_t)node * HIDC + c]);
  }
  if (lane == 0) atomicAdd(&cnt[gidx], 1.0f);
}

__global__ __launch_bounds__(256) void pool_final(float* featsum, const float* cnt) {
  int i = blockIdx.x * 256 + threadIdx.x;        // < 64*128
  if (i < NGRAPH * HIDC) featsum[i] /= fmaxf(cnt[i >> 7], 1.0f);
}

// ----------------------- head row-norm + ReLU (in place) -------------------
// One 256-thread block per row; LDS cross-wave reduction.
__global__ __launch_bounds__(256) void rownorm_relu(float* __restrict__ z, int D) {
  __shared__ float sred[8];
  __shared__ float smu, srs;
  int row = blockIdx.x, tid = threadIdx.x, lane = tid & 31, w = tid >> 5;
  float* rp = z + (size_t)row * D;

  float s = 0.0f;
  for (int c = tid; c < D; c += 256) s += rp[c];
  #pragma unroll
  for (int o = 16; o >= 1; o >>= 1) s += __shfl_xor(s, o, 32);
  if (lane == 0) sred[w] = s;
  __syncthreads();
  if (tid == 0) { float t = 0; for (int i = 0; i < 8; ++i) t += sred[i]; smu = t / D; }
  __syncthreads();
  float mu = smu;

  float vs = 0.0f;
  for (int c = tid; c < D; c += 256) { float d = rp[c] - mu; vs += d * d; }
  #pragma unroll
  for (int o = 16; o >= 1; o >>= 1) vs += __shfl_xor(vs, o, 32);
  __syncthreads();
  if (lane == 0) sred[w] = vs;
  __syncthreads();
  if (tid == 0) { float t = 0; for (int i = 0; i < 8; ++i) t += sred[i];
                  srs = rsqrtf(t / D + 1e-5f); }
  __syncthreads();
  float rs = srs;
  for (int c = tid; c < D; c += 256)
    rp[c] = fmaxf((rp[c] - mu) * rs, 0.0f);       // ReLU after row-norm
}

// Final tiny GEMM: out[64,4] = z2[64,256] @ fcW3[256,4] + fcb3
__global__ __launch_bounds__(256) void head_final(
    const float* __restrict__ z2, const float* __restrict__ W,
    const float* __restrict__ b, float* __restrict__ out) {
  int i = blockIdx.x * 256 + threadIdx.x;
  if (i >= NGRAPH * 4) return;
  int r = i >> 2, c = i & 3;
  float s = b[c];
  for (int k = 0; k < 256; ++k) s += z2[(size_t)r * 256 + k] * W[(size_t)k * 4 + c];
  out[i] = s;
}

// ------------------------------- driver ------------------------------------

extern "C" void kernel_launch(void* const* d_in, const int* in_sizes, int n_in,
                              void* d_out, int out_size, void* d_ws, size_t ws_size,
                              hipStream_t stream) {
  const float*     x     = (const float*)d_in[0];
  const long long* ei    = (const long long*)d_in[1];
  const long long* batch = (const long long*)d_in[2];
  const float* W[4]  = {(const float*)d_in[3],  (const float*)d_in[7],
                        (const float*)d_in[11], (const float*)d_in[15]};
  const float* bb[4] = {(const float*)d_in[4],  (const float*)d_in[8],
                        (const float*)d_in[12], (const float*)d_in[16]};
  const float* gg[4] = {(const float*)d_in[5],  (const float*)d_in[9],
                        (const float*)d_in[13], (const float*)d_in[17]};
  const float* be[4] = {(const float*)d_in[6],  (const float*)d_in[10],
                        (const float*)d_in[14], (const float*)d_in[18]};
  const float* fcW1 = (const float*)d_in[19]; const float* fcb1 = (const float*)d_in[20];
  const float* fcW2 = (const float*)d_in[21]; const float* fcb2 = (const float*)d_in[22];
  const float* fcW3 = (const float*)d_in[23]; const float* fcb3 = (const float*)d_in[24];
  float* out = (float*)d_out;

  const int N = in_sizes[0] / 4;      // 100000
  const int E = in_sizes[1] / 2;      // 1600000

  // carve workspace
  size_t off = 0;
  auto carve = [&](size_t bytes) -> float* {
    float* p = (float*)((char*)d_ws + off);
    off += (bytes + 255) & ~(size_t)255;
    return p;
  };
  float* deg  = carve((size_t)N * 4);
  float* dinv = carve((size_t)N * 4);
  float* hA   = carve((size_t)N * HIDC * 4);   // current node features
  float* hB   = carve((size_t)N * HIDC * 4);   // h @ W
  float* hC   = carve((size_t)N * HIDC * 4);   // edge aggregate
  float* feat = carve((size_t)NGRAPH * HIDC * 4);
  float* cnt  = carve((size_t)NGRAPH * 4);
  float* z1   = carve((size_t)NGRAPH * 1024 * 4);
  float* z2   = carve((size_t)NGRAPH * 256 * 4);
  (void)ws_size; (void)n_in; (void)out_size;

  const int nBlk   = (N + 255) / 256;
  const int eBlk   = (E + 255) / 256;
  const int wNode  = (N + 7) / 8;              // 8 waves/block, wave per node
  const long long edgeThreads = (long long)E * 32;
  const int eScBlk = (int)((edgeThreads + 255) / 256);
  const long long nodeThreads = (long long)N * 32;
  const int nScBlk = (int)((nodeThreads + 255) / 256);

  // degrees
  deg_init<<<nBlk, 256, 0, stream>>>(deg, N);
  deg_accum<<<eBlk, 256, 0, stream>>>(ei, deg, E);
  deg_inv<<<nBlk, 256, 0, stream>>>(deg, dinv, N);

  // 4 GCN layers
  for (int l = 0; l < 4; ++l) {
    if (l == 0) {
      gemm_k4<<<nScBlk, 256, 0, stream>>>(x, W[0], hB, N);
    } else {
      int tiles = (N / 16) * (HIDC / 16);
      wmma_gemm_bf16<<<(tiles + 7) / 8, 256, 0, stream>>>(hA, W[l], nullptr, hB,
                                                          N, HIDC, HIDC);
    }
    zero_f32<<<2048, 256, 0, stream>>>(hC, (long long)N * HIDC);
    scatter_edges<<<eScBlk, 256, 0, stream>>>(ei, hB, dinv, hC, E);
    gcn_post<<<wNode, 256, 0, stream>>>(hC, hB, dinv, bb[l], gg[l], be[l], hA, N);
  }

  // mean pool
  zero_f32<<<64, 256, 0, stream>>>(feat, (long long)NGRAPH * HIDC);
  zero_f32<<<1, 64, 0, stream>>>(cnt, NGRAPH);
  pool_accum<<<wNode, 256, 0, stream>>>(hA, batch, feat, cnt, N);
  pool_final<<<(NGRAPH * HIDC + 255) / 256, 256, 0, stream>>>(feat, cnt);

  // head
  {
    int tiles = (NGRAPH / 16) * (1024 / 16);    // 256 tiles
    wmma_gemm_bf16<<<(tiles + 7) / 8, 256, 0, stream>>>(feat, fcW1, fcb1, z1,
                                                        NGRAPH, 1024, HIDC);
    rownorm_relu<<<NGRAPH, 256, 0, stream>>>(z1, 1024);
    tiles = (NGRAPH / 16) * (256 / 16);         // 64 tiles
    wmma_gemm_bf16<<<(tiles + 7) / 8, 256, 0, stream>>>(z1, fcW2, fcb2, z2,
                                                        NGRAPH, 256, 1024);
    rownorm_relu<<<NGRAPH, 256, 0, stream>>>(z2, 256);
    head_final<<<1, 256, 0, stream>>>(z2, fcW3, fcb3, out);
  }
}
